// Lstm_70884140253875
// MI455X (gfx1250) — matrix operand order repaired and checked by reading, not compiled, not run on hardware
//
#include <hip/hip_runtime.h>
#include <hip/hip_bf16.h>
#include <stdint.h>

typedef __bf16 bf16_t;
typedef bf16_t v16bf __attribute__((ext_vector_type(16)));
typedef float  v8f   __attribute__((ext_vector_type(8)));

union FragU {
    v16bf v;
    uint4 u[2];
};

__device__ __forceinline__ unsigned short f2bf(float f) {
    unsigned int u = __float_as_uint(f);
    unsigned int r = u + 0x7fffu + ((u >> 16) & 1u);  // round-to-nearest-even
    return (unsigned short)(r >> 16);
}

__device__ __forceinline__ float sigmoidf_(float x) {
    return 1.0f / (1.0f + __expf(-x));
}

// ---------------------------------------------------------------------------
// f32 -> bf16 conversion (grid-stride, flat layout) -- used for A matrices
// ---------------------------------------------------------------------------
__global__ __launch_bounds__(256) void f32_to_bf16_kernel(const float* __restrict__ in,
                                                          unsigned short* __restrict__ out,
                                                          int n) {
    int i = blockIdx.x * blockDim.x + threadIdx.x;
    int stride = gridDim.x * blockDim.x;
    for (; i < n; i += stride) out[i] = f2bf(in[i]);
}

// ---------------------------------------------------------------------------
// f32 [K,N] row-major -> bf16 panel-swizzled [K/32][N][32] (K contiguous per
// column within a 32-wide K panel). One thread per (panel, column).
// Reads coalesced over n; each thread writes one contiguous 64B column.
// ---------------------------------------------------------------------------
__global__ __launch_bounds__(256) void f32_to_bf16_swz_kernel(const float* __restrict__ in,
                                                              unsigned short* __restrict__ out,
                                                              int K, int N) {
    int idx = blockIdx.x * blockDim.x + threadIdx.x;
    int total = (K >> 5) * N;
    if (idx >= total) return;
    int p = idx / N, n = idx - p * N;
    unsigned short tmp[32];
#pragma unroll
    for (int kk = 0; kk < 32; kk++)
        tmp[kk] = f2bf(in[(size_t)(p * 32 + kk) * N + n]);
    uint4* dst = reinterpret_cast<uint4*>(out + ((size_t)p * N + n) * 32);
    const uint4* src = reinterpret_cast<const uint4*>(tmp);
    dst[0] = src[0]; dst[1] = src[1]; dst[2] = src[2]; dst[3] = src[3];
}

// ---------------------------------------------------------------------------
// initial state: c0,h0 = initial_state.reshape(B,T,2H)[:,0], masked by mask[0]
// ---------------------------------------------------------------------------
__global__ __launch_bounds__(256) void init_state_kernel(const float* __restrict__ is,
                                                         const float* __restrict__ mask,
                                                         float* __restrict__ c,
                                                         unsigned short* __restrict__ h_bf,
                                                         int B, int T, int H) {
    int idx = blockIdx.x * blockDim.x + threadIdx.x;
    if (idx >= B * H) return;
    int b = idx / H, j = idx - b * H;
    const float* row = is + (size_t)b * T * 2 * H;  // row (b*T) of (B*T, 2H)
    float keep = 1.0f - mask[b];                    // mask[t=0, b]
    c[idx] = row[j] * keep;
    h_bf[idx] = f2bf(row[H + j] * keep);
}

// ---------------------------------------------------------------------------
// bf16 WMMA GEMM: Cpart[z][M,N] = A[M, kBase:kBase+kLen] * Bsw[kBase:.., N]
// A row-major; B panel-swizzled [K/32][N][32]. Block tile 128x128, 8 wave32
// (4 in M x 2 in N), wave tile 32x64 = 2x4 WMMA tiles, double-buffered LDS,
// 8 v_wmma_f32_16x16x32_bf16 per K-step. Split-K across blockIdx.z.
// ---------------------------------------------------------------------------
#define BM 128
#define BN 128
#define BK 32
#define AK 40  // padded LDS row stride (ushorts): conflict-free b128 reads

__global__ __launch_bounds__(256) void gemm_bf16_kernel(const unsigned short* __restrict__ A,
                                                        const unsigned short* __restrict__ Bsw,
                                                        float* __restrict__ C,
                                                        int M, int N, int kLen) {
    __shared__ __align__(16) unsigned short As[2][BM][AK];   // A tiles, row-major
    __shared__ __align__(16) unsigned short Bts[2][BN][AK];  // B tiles, n-major (K contiguous)

    const int tid   = threadIdx.x;
    const int m0    = blockIdx.y * BM;
    const int n0    = blockIdx.x * BN;
    const int kBase = blockIdx.z * kLen;
    float* Cp       = C + (size_t)blockIdx.z * M * N;

    const int wid  = tid >> 5;
    const int lane = tid & 31;
    const int wm   = (wid & 3) * 32;   // wave row base in tile
    const int wn   = (wid >> 2) * 64;  // wave col base in tile
    const int lr   = lane & 15;        // position within 16
    const int lh   = lane >> 4;        // lane half

    v8f acc[2][4];
#pragma unroll
    for (int mi = 0; mi < 2; mi++)
#pragma unroll
        for (int ni = 0; ni < 4; ni++) acc[mi][ni] = (v8f)0.0f;

    // staging coords (A and B staging have identical shape now)
    const int arow = tid >> 1;          // 0..127
    const int akh  = (tid & 1) * 16;    // 0,16

    // A row stride = total K of the problem
    const int K = kLen * gridDim.z;

    uint4 ga0, ga1, gb0, gb1;
    auto loadRegsK = [&](int k0) {
        const uint4* ga = reinterpret_cast<const uint4*>(A + (size_t)(m0 + arow) * K + k0 + akh);
        ga0 = ga[0];
        ga1 = ga[1];
        const uint4* gb = reinterpret_cast<const uint4*>(
            Bsw + ((size_t)(k0 >> 5) * N + n0 + arow) * 32 + akh);
        gb0 = gb[0];
        gb1 = gb[1];
    };
    auto storeLds = [&](int buf) {
        *reinterpret_cast<uint4*>(&As[buf][arow][akh])      = ga0;
        *reinterpret_cast<uint4*>(&As[buf][arow][akh + 8])  = ga1;
        *reinterpret_cast<uint4*>(&Bts[buf][arow][akh])     = gb0;
        *reinterpret_cast<uint4*>(&Bts[buf][arow][akh + 8]) = gb1;
    };

    const int kEnd = kBase + kLen;
    loadRegsK(kBase);
    storeLds(0);
    __syncthreads();

    int buf = 0;
    for (int k0 = kBase; k0 < kEnd; k0 += BK) {
        int kn = k0 + BK;
        bool more = kn < kEnd;
        if (more) {
            loadRegsK(kn);  // issue global loads; latency covered by WMMAs below
            __builtin_prefetch(Bsw + ((size_t)(kn >> 5) * N + n0 + arow) * 32 + akh, 0, 1);
        }

        // ---- fragments from LDS (layouts per ISA 7.12.2)
        FragU aF[2], bF[4];
#pragma unroll
        for (int mi = 0; mi < 2; mi++) {
            int r = wm + mi * 16 + lr;
            int kb = lh * 8;
            aF[mi].u[0] = *reinterpret_cast<const uint4*>(&As[buf][r][kb]);        // K kb..kb+7
            aF[mi].u[1] = *reinterpret_cast<const uint4*>(&As[buf][r][kb + 16]);   // K kb+16..kb+23
        }
#pragma unroll
        for (int ni = 0; ni < 4; ni++) {
            int cc = wn + ni * 16 + lr;
            int kb = lh * 8;
            bF[ni].u[0] = *reinterpret_cast<const uint4*>(&Bts[buf][cc][kb]);      // K kb..kb+7
            bF[ni].u[1] = *reinterpret_cast<const uint4*>(&Bts[buf][cc][kb + 16]); // K kb+16..kb+23
        }
#pragma unroll
        for (int mi = 0; mi < 2; mi++)
#pragma unroll
            for (int ni = 0; ni < 4; ni++)
                acc[mi][ni] = __builtin_amdgcn_wmma_f32_16x16x32_bf16(
                    false, aF[mi].v, false, bF[ni].v, (short)0, acc[mi][ni], false, false);

        if (more) storeLds(buf ^ 1);  // after compute: wait lands past the WMMA block
        __syncthreads();
        buf ^= 1;
    }

    // ---- epilogue: C/D layout: row = base + lh*8 + v, col = base + lr
#pragma unroll
    for (int mi = 0; mi < 2; mi++)
#pragma unroll
        for (int ni = 0; ni < 4; ni++)
#pragma unroll
            for (int v = 0; v < 8; v++) {
                int r = m0 + wm + mi * 16 + lh * 8 + v;
                int c = n0 + wn + ni * 16 + lr;
                Cp[(size_t)r * N + c] = acc[mi][ni][v];
            }
}

// ---------------------------------------------------------------------------
// block-wide sum over 256 threads (wave32 shuffles + LDS)
// ---------------------------------------------------------------------------
__device__ __forceinline__ float blockSum256(float v, float* sm) {
    int lane = threadIdx.x & 31, wid = threadIdx.x >> 5;
#pragma unroll
    for (int o = 16; o; o >>= 1) v += __shfl_down(v, o, 32);
    if (lane == 0) sm[wid] = v;
    __syncthreads();
    if (wid == 0) {
        float t = (lane < 8) ? sm[lane] : 0.0f;
#pragma unroll
        for (int o = 4; o; o >>= 1) t += __shfl_down(t, o, 32);
        if (lane == 0) sm[0] = t;
    }
    __syncthreads();
    float r = sm[0];
    __syncthreads();
    return r;
}

// ---------------------------------------------------------------------------
// Row LayerNorm over 8192 cols, in place: z = LN(z)*g + b. One block per row.
// ---------------------------------------------------------------------------
__global__ __launch_bounds__(256) void ln_rows_kernel(float* __restrict__ z,
                                                      const float* __restrict__ g,
                                                      const float* __restrict__ bvec) {
    __shared__ float sm[8];
    float* row = z + (size_t)blockIdx.x * 8192;
    const int tid = threadIdx.x;
    float v[32];
    float s1 = 0.0f, s2 = 0.0f;
#pragma unroll
    for (int k = 0; k < 32; k++) {
        float x = row[tid + k * 256];
        v[k] = x; s1 += x; s2 += x * x;
    }
    float tot  = blockSum256(s1, sm);
    float tot2 = blockSum256(s2, sm);
    float mean = tot * (1.0f / 8192.0f);
    float var  = tot2 * (1.0f / 8192.0f) - mean * mean;
    float rstd = rsqrtf(var + 1e-5f);
#pragma unroll
    for (int k = 0; k < 32; k++) {
        int j = tid + k * 256;
        row[j] = (v[k] - mean) * rstd * g[j] + bvec[j];
    }
}

// ---------------------------------------------------------------------------
// Fused LSTM step: sums split-K partials of zh, LN(zh), gates, cell update,
// LN(c), h; writes hs[t]; applies mask[t+1] to stored state; writes s at end.
// One block per batch row. 4H = 8192, H = 2048.
// ---------------------------------------------------------------------------
__global__ __launch_bounds__(256) void lstm_step_kernel(const float* __restrict__ zh,   // 2 planes of [B][8192]
                                                        const float* __restrict__ zxln,
                                                        const float* __restrict__ bias,
                                                        const float* __restrict__ gh,
                                                        const float* __restrict__ bh,
                                                        const float* __restrict__ gc,
                                                        const float* __restrict__ bc,
                                                        const float* __restrict__ mask,
                                                        int t, int T, int B,
                                                        float* __restrict__ c,
                                                        unsigned short* __restrict__ h_bf,
                                                        float* __restrict__ hs_out,
                                                        float* __restrict__ s_out) {
    __shared__ float sm[8];
    const int b = blockIdx.x, tid = threadIdx.x;
    const int H = 2048;
    const float* zrow0 = zh + (size_t)b * 8192;
    const float* zrow1 = zh + (size_t)B * 8192 + (size_t)b * 8192;
    const float* xrow  = zxln + (size_t)b * 8192;

    // pass 1: sum split-K partials, accumulate stats
    float v[32];
    float s1 = 0.0f, s2 = 0.0f;
#pragma unroll
    for (int k = 0; k < 32; k++) {
        int j = tid + k * 256;
        float z = zrow0[j] + zrow1[j];
        v[k] = z; s1 += z; s2 += z * z;
    }
    float tot  = blockSum256(s1, sm);
    float tot2 = blockSum256(s2, sm);
    float mean = tot * (1.0f / 8192.0f);
    float var  = tot2 * (1.0f / 8192.0f) - mean * mean;
    float rstd = rsqrtf(var + 1e-5f);

    // z = LN(zh)*gh + bh + zx_ln + b
#pragma unroll
    for (int k = 0; k < 32; k++) {
        int j = tid + k * 256;
        v[k] = (v[k] - mean) * rstd * gh[j] + bh[j] + xrow[j] + bias[j];
    }

    // gates: thread owns j = tid + k*256 for k=0..7 (i/f/o/u at k, k+8, k+16, k+24)
    float cn[8], oo[8];
    float cs1 = 0.0f, cs2 = 0.0f;
#pragma unroll
    for (int k = 0; k < 8; k++) {
        int j = tid + k * 256;
        float ig = sigmoidf_(v[k]);
        float fg = sigmoidf_(v[k + 8]);
        float og = sigmoidf_(v[k + 16]);
        float ug = tanhf(v[k + 24]);
        float cc = fg * c[(size_t)b * H + j] + ig * ug;
        cn[k] = cc; oo[k] = og;
        cs1 += cc; cs2 += cc * cc;
    }
    float ct  = blockSum256(cs1, sm);
    float ct2 = blockSum256(cs2, sm);
    float mc = ct * (1.0f / 2048.0f);
    float vc = ct2 * (1.0f / 2048.0f) - mc * mc;
    float rc = rsqrtf(vc + 1e-5f);

    float mnext = (t + 1 < T) ? mask[(size_t)(t + 1) * B + b] : 0.0f;
    float keep  = 1.0f - mnext;
    bool  last  = (t == T - 1);

#pragma unroll
    for (int k = 0; k < 8; k++) {
        int j = tid + k * 256;
        float hh = oo[k] * tanhf((cn[k] - mc) * rc * gc[j] + bc[j]);
        hs_out[(size_t)b * H + j] = hh;                 // unmasked output
        c[(size_t)b * H + j]   = cn[k] * keep;          // state masked for next step
        h_bf[(size_t)b * H + j] = f2bf(hh * keep);
        if (last) {
            s_out[(size_t)b * 2 * H + j]     = cn[k];   // final state unmasked
            s_out[(size_t)b * 2 * H + H + j] = hh;
        }
    }
}

// ---------------------------------------------------------------------------
extern "C" void kernel_launch(void* const* d_in, const int* in_sizes, int n_in,
                              void* d_out, int out_size, void* d_ws, size_t ws_size,
                              hipStream_t stream) {
    const float* x    = (const float*)d_in[0];
    const float* mask = (const float*)d_in[1];
    const float* istate = (const float*)d_in[2];
    const float* wx   = (const float*)d_in[3];
    const float* wh   = (const float*)d_in[4];
    const float* bvec = (const float*)d_in[5];
    const float* gx   = (const float*)d_in[6];
    const float* bx   = (const float*)d_in[7];
    const float* gh   = (const float*)d_in[8];
    const float* bh   = (const float*)d_in[9];
    const float* gc   = (const float*)d_in[10];
    const float* bc   = (const float*)d_in[11];

    const int T = 64, B = 256, NIN = 1024, H = 2048;
    const int FH = 4 * H;  // 8192

    char* ws = (char*)d_ws;
    size_t off = 0;
    auto alloc = [&](size_t bytes) -> void* {
        void* p = ws + off;
        off = (off + bytes + 255) & ~(size_t)255;
        return p;
    };
    unsigned short* wx_bf = (unsigned short*)alloc((size_t)NIN * FH * 2);  // swizzled
    unsigned short* wh_bf = (unsigned short*)alloc((size_t)H * FH * 2);    // swizzled
    unsigned short* x_bf  = (unsigned short*)alloc((size_t)T * B * NIN * 2);
    float* zx   = (float*)alloc((size_t)T * B * FH * 4);
    float* zh   = (float*)alloc((size_t)2 * B * FH * 4);  // 2 split-K planes
    float* cbuf = (float*)alloc((size_t)B * H * 4);
    unsigned short* hbf = (unsigned short*)alloc((size_t)B * H * 2);

    float* out  = (float*)d_out;
    float* hs   = out;                        // (T, B, H)
    float* sout = out + (size_t)T * B * H;    // (B, 2H)

    // setup: bf16 weights (panel-swizzled) / inputs + masked initial state
    f32_to_bf16_swz_kernel<<<((NIN / 32) * FH + 255) / 256, 256, 0, stream>>>(wx, wx_bf, NIN, FH);
    f32_to_bf16_swz_kernel<<<((H / 32) * FH + 255) / 256, 256, 0, stream>>>(wh, wh_bf, H, FH);
    f32_to_bf16_kernel<<<8192, 256, 0, stream>>>(x, x_bf, T * B * NIN);
    init_state_kernel<<<(B * H + 255) / 256, 256, 0, stream>>>(istate, mask, cbuf, hbf, B, T, H);

    // phase A: ZX = X @ wx for all timesteps (split=1), then row LN (gx, bx)
    gemm_bf16_kernel<<<dim3(FH / BN, (T * B) / BM, 1), 256, 0, stream>>>(x_bf, wx_bf, zx, T * B, FH, NIN);
    ln_rows_kernel<<<T * B, 256, 0, stream>>>(zx, gx, bx);

    // phase B: recurrence, split-K=2 GEMM (two partial planes of zh)
    for (int t = 0; t < T; t++) {
        gemm_bf16_kernel<<<dim3(FH / BN, B / BM, 2), 256, 0, stream>>>(hbf, wh_bf, zh, B, FH, H / 2);
        lstm_step_kernel<<<B, 256, 0, stream>>>(zh, zx + (size_t)t * B * FH, bvec,
                                                gh, bh, gc, bc, mask, t, T, B,
                                                cbuf, hbf, hs + (size_t)t * B * H, sout);
    }
}
